// SparseSAKELayer_68917045232002
// MI455X (gfx1250) — matrix operand order repaired and checked by reading, not compiled, not run on hardware
//
#include <hip/hip_runtime.h>

#define F   32
#define H   32
#define NHD 4
#define NC  128
#define KFN 50
#define CUT 5.0f

typedef _Float16 h16;
typedef __attribute__((ext_vector_type(16))) _Float16 v16h;
typedef __attribute__((ext_vector_type(8)))  float    v8f;

// ---------------------------------------------------------------------------
// wave-local LDS fence: all prior ds ops complete before later ones issue.
__device__ inline void wave_lds_wait() {
  asm volatile("s_wait_dscnt 0" ::: "memory");
}

__device__ inline float siluf(float x) { return x / (1.0f + __expf(-x)); }
__device__ inline float celu2f(float x) {
  return fmaxf(x, 0.0f) + fminf(0.0f, 2.0f * (__expf(x * 0.5f) - 1.0f));
}
// order-preserving float<->uint encoding for atomicMax on floats
__device__ inline unsigned fenc(float f) {
  unsigned u = __float_as_uint(f);
  return (f >= 0.0f) ? (u | 0x80000000u) : ~u;
}
__device__ inline float fdec(unsigned u) {
  return (u & 0x80000000u) ? __uint_as_float(u & 0x7fffffffu)
                           : __uint_as_float(~u);
}
#define AMAX_NEG_INF 0x007FFFFFu  // fenc(-inf)

// ---------------------------------------------------------------------------
// Fragment-swizzled LDS layouts (wave32 WMMA f16 16x16x32).
// Each 16x32(A) or 32x16(B) fragment chunk = 512 halves; lane's 16 halves
// are contiguous at chunk*512 + lane*16.
//
// A element (m, k): lane = m + 16*((k>>3)&1), elem = (k&7) + ((k&16)?8:0)
__device__ inline int a_idx(int m, int k) {
  int kk = k & 31;
  return ((k >> 5) << 9) + ((m + ((kk & 8) << 1)) << 4) + (kk & 7) +
         ((kk & 16) ? 8 : 0);
}
// forward map for fragment-order producers: k for (lane, chunk c, elem e)
__device__ inline int a_k_of(int lane, int c, int e) {
  return (c << 5) + ((lane & 16) ? 8 : 0) + (e & 7) + ((e & 8) ? 16 : 0);
}
// B element (k, n) for a matrix with NT column tiles:
template <int NT>
__device__ inline int b_idx(int k, int n) {
  return (((k >> 5) * NT + (n >> 4)) << 9) +
         (((n & 15) + ((k & 8) << 1)) << 4) + (k & 7) + ((k & 16) ? 8 : 0);
}

template <int SZ>
__device__ inline v16h frag_ld(const h16 (&P)[SZ], int off) {
  return *(const v16h*)&P[off];
}
template <int SZ>
__device__ inline void frag_st(h16 (&P)[SZ], int off, v16h v) {
  *(v16h*)&P[off] = v;
}

// D(16xN) = A(16xK) @ B(KxN); epilogue epi(nt, acc) applied per 16x16 tile.
template <int K, int N, int SZA, int SZB, typename Epi>
__device__ inline void wave_gemm(const h16 (&A)[SZA], int abase,
                                 const h16 (&B)[SZB], int lane, Epi epi) {
  constexpr int NT = N / 16;
  constexpr int KC = K / 32;
  v16h af[KC];
#pragma unroll
  for (int c = 0; c < KC; ++c)
    af[c] = frag_ld(A, abase + (c << 9) + lane * 16);
#pragma unroll
  for (int nt = 0; nt < NT; ++nt) {
    v8f acc = {};
#pragma unroll
    for (int c = 0; c < KC; ++c) {
      v16h b = frag_ld(B, ((c * NT + nt) << 9) + lane * 16);
      acc = __builtin_amdgcn_wmma_f32_16x16x32_f16(false, af[c], false, b,
                                                   (short)0, acc, false, false);
    }
    epi(nt, acc);
  }
}

// ---------------------------------------------------------------------------
__global__ void k_init(unsigned* amax, float* zbase, long long zcount,
                       int amaxCount) {
  long long i      = (long long)blockIdx.x * blockDim.x + threadIdx.x;
  long long stride = (long long)gridDim.x * blockDim.x;
  for (long long j = i; j < zcount; j += stride) zbase[j] = 0.0f;
  for (long long j = i; j < amaxCount; j += stride) amax[j] = AMAX_NEG_INF;
}

// ---------------------------------------------------------------------------
// Pass 1 (edges): he = silu(cat@We1+b)@We2+b ; att = celu(he@Watt+b)
__global__ __launch_bounds__(64) void k_edge1(
    const float* __restrict__ h, const float* __restrict__ x,
    const int* __restrict__ edges, const float* W_in, const float* b_in,
    const float* W_e1, const float* b_e1, const float* W_e2, const float* b_e2,
    const float* W_att, const float* b_att, float* he_out, float* attc_out,
    unsigned* amax, float* cnt, int E) {
  __shared__ __align__(32) h16 sWin[2 * 4 * 512];   // K64  x N64
  __shared__ __align__(32) h16 sWe1[4 * 2 * 512];   // K128 x N32
  __shared__ __align__(32) h16 sWe2[1 * 2 * 512];   // K32  x N32
  __shared__ __align__(32) h16 sWatt[1 * 1 * 512];  // K32  x N16
  __shared__ float sbin[64], sbe1[32], sbe2[32], sbatt[16];
  __shared__ __align__(32) h16 sCat[2][4 * 512];  // K128 A
  __shared__ __align__(32) h16 sAb[2][512];
  __shared__ __align__(32) h16 sAb2[2][512];
  __shared__ float sdv[2][16];
  __shared__ int   stgt[2][16], ssrc[2][16];

  int tid = threadIdx.x;
  for (int i = tid; i < 64 * 64; i += blockDim.x) {
    int k = i >> 6, n = i & 63;
    sWin[b_idx<4>(k, n)] = (h16)((n < KFN) ? W_in[k * KFN + n] : 0.0f);
  }
  for (int i = tid; i < 128 * 32; i += blockDim.x) {
    int k = i >> 5, n = i & 31;
    sWe1[b_idx<2>(k, n)] = (h16)((k < 115) ? W_e1[k * 32 + n] : 0.0f);
  }
  for (int i = tid; i < 32 * 32; i += blockDim.x) {
    int k = i >> 5, n = i & 31;
    sWe2[b_idx<2>(k, n)] = (h16)W_e2[i];
  }
  for (int i = tid; i < 32 * 16; i += blockDim.x) {
    int k = i >> 4, n = i & 15;
    sWatt[b_idx<1>(k, n)] = (h16)((n < NHD) ? W_att[k * NHD + n] : 0.0f);
  }
  for (int i = tid; i < 64; i += blockDim.x) sbin[i] = (i < KFN) ? b_in[i] : 0.0f;
  for (int i = tid; i < 32; i += blockDim.x) { sbe1[i] = b_e1[i]; sbe2[i] = b_e2[i]; }
  for (int i = tid; i < 16; i += blockDim.x) sbatt[i] = (i < NHD) ? b_att[i] : 0.0f;
  __syncthreads();

  int w = tid >> 5, lane = tid & 31;
  int nl = lane & 15, mb = (lane & 16) ? 8 : 0;
  int wpb = blockDim.x >> 5;
  int wid = blockIdx.x * wpb + w;
  int wstride = gridDim.x * wpb;
  int tiles = (E + 15) >> 4;

  const float m0    = __expf(-CUT);
  const float dmu   = (1.0f - m0) / (float)(KFN - 1);
  const float bt0   = (2.0f / (float)KFN) * (1.0f - m0);
  const float betas = 1.0f / (bt0 * bt0);

  for (int t = wid; t < tiles; t += wstride) {
    int e0 = t << 4;
    if (lane < 16) {
      int ed = e0 + lane;
      int s = 0, g = 0;
      float dd = 0.0f;
      if (ed < E) {
        s = edges[2 * ed];
        g = edges[2 * ed + 1];
        float dx0 = x[3 * s + 0] - x[3 * g + 0];
        float dx1 = x[3 * s + 1] - x[3 * g + 1];
        float dx2 = x[3 * s + 2] - x[3 * g + 2];
        dd = sqrtf(dx0 * dx0 + dx1 * dx1 + dx2 * dx2 + 1e-14f);
      }
      ssrc[w][lane] = s;
      stgt[w][lane] = g;
      sdv[w][lane]  = dd;
    }
    wave_lds_wait();
    // cat chunks 0..1 = [h[src], h[tgt]] (fragment-major build)
    {
      int m = lane & 15;
      int ed = e0 + m;
#pragma unroll
      for (int c = 0; c < 2; ++c) {
        v16h val;
#pragma unroll
        for (int e = 0; e < 16; ++e) {
          int k = a_k_of(lane, c, e);
          float vv = 0.0f;
          if (ed < E) {
            int node = (k < F) ? ssrc[w][m] : stgt[w][m];
            vv = h[node * F + (k & 31)];
          }
          val[e] = (h16)vv;
        }
        frag_st(sCat[w], (c << 9) + lane * 16, val);
      }
    }
    wave_lds_wait();
    // h1 = hh @ W_in + b_in ; cat[:,64:115] = [rbf*h1, d] (chunks 2..3)
    wave_gemm<64, 64>(sCat[w], 0, sWin, lane, [&](int nt, v8f acc) {
      int n = (nt << 4) + nl;
#pragma unroll
      for (int r = 0; r < 8; ++r) {
        int m = mb + r;
        float hv = acc[r] + sbin[n];
        float vv;
        if (n < KFN) {
          float dd  = sdv[w][m];
          float edv = __expf(-dd);
          float mu  = m0 + (float)n * dmu;
          float df  = edv - mu;
          vv = __expf(-betas * df * df) * hv;
        } else if (n == KFN) {
          vv = sdv[w][m];
        } else {
          vv = 0.0f;
        }
        sCat[w][a_idx(m, 64 + n)] = (h16)vv;
      }
    });
    wave_lds_wait();
    wave_gemm<128, 32>(sCat[w], 0, sWe1, lane, [&](int nt, v8f acc) {
      int n = (nt << 4) + nl;
#pragma unroll
      for (int r = 0; r < 8; ++r)
        sAb[w][a_idx(mb + r, n)] = (h16)siluf(acc[r] + sbe1[n]);
    });
    wave_lds_wait();
    wave_gemm<32, 32>(sAb[w], 0, sWe2, lane, [&](int nt, v8f acc) {
      int n = (nt << 4) + nl;
#pragma unroll
      for (int r = 0; r < 8; ++r) {
        int m = mb + r;
        int ed = e0 + m;
        float vv = acc[r] + sbe2[n];
        if (ed < E) he_out[(size_t)ed * 32 + n] = vv;
        sAb2[w][a_idx(m, n)] = (h16)vv;
      }
    });
    wave_lds_wait();
    wave_gemm<32, 16>(sAb2[w], 0, sWatt, lane, [&](int nt, v8f acc) {
      (void)nt;
      if (nl < NHD) {
#pragma unroll
        for (int r = 0; r < 8; ++r) {
          int m = mb + r;
          int ed = e0 + m;
          if (ed < E) {
            float a = celu2f(acc[r] + sbatt[nl]);
            attc_out[(size_t)ed * NHD + nl] = a;
            atomicMax(&amax[stgt[w][m] * NHD + nl], fenc(a));
          }
        }
      }
    });
    if (lane < 16) {
      int ed = e0 + lane;
      if (ed < E) atomicAdd(&cnt[stgt[w][lane]], 1.0f);
    }
  }
}

// ---------------------------------------------------------------------------
// Pass 2: segment sum of exp(att - max)
__global__ void k_edge2(const float* __restrict__ attc,
                        const unsigned* __restrict__ amax,
                        const int* __restrict__ edges, float* asum, int E) {
  int e = blockIdx.x * blockDim.x + threadIdx.x;
  if (e >= E) return;
  int g = edges[2 * e + 1];
#pragma unroll
  for (int c = 0; c < NHD; ++c) {
    float a  = attc[(size_t)e * NHD + c];
    float mx = fdec(amax[g * NHD + c]);
    atomicAdd(&asum[g * NHD + c], __expf(a - mx));
  }
}

// ---------------------------------------------------------------------------
// Pass 3: coeff = tanh(outer(he, comb) @ W_x); scatter h_e, comb_sum, dv.
__global__ __launch_bounds__(64) void k_edge3(
    const float* __restrict__ x, const int* __restrict__ edges,
    const float* __restrict__ he, const float* __restrict__ attc,
    const unsigned* __restrict__ amax, const float* __restrict__ asum,
    const float* W_x, const float* w_vmix, float* h_e, float* csum,
    float* dvacc, int E) {
  __shared__ __align__(32) h16 sWx[4 * 8 * 512];  // K128 x N128
  __shared__ float svmix[128];
  __shared__ __align__(32) h16 sA[2][4 * 512];
  __shared__ float scomb[2][16 * NHD];
  __shared__ float sdxn[2][16 * 3];
  __shared__ float svm[2][16];
  __shared__ int   stgt[2][16];

  int tid = threadIdx.x;
  for (int i = tid; i < 128 * 128; i += blockDim.x) {
    int k = i >> 7, n = i & 127;
    sWx[b_idx<8>(k, n)] = (h16)W_x[i];
  }
  for (int i = tid; i < 128; i += blockDim.x) svmix[i] = w_vmix[i];
  __syncthreads();

  int w = tid >> 5, lane = tid & 31;
  int nl = lane & 15, mb = (lane & 16) ? 8 : 0;
  int wpb = blockDim.x >> 5;
  int wid = blockIdx.x * wpb + w;
  int wstride = gridDim.x * wpb;
  int tiles = (E + 15) >> 4;

  for (int t = wid; t < tiles; t += wstride) {
    int e0 = t << 4;
    if (lane < 16) {
      int ed = e0 + lane;
      int g = 0;
      float d0 = 0.f, d1 = 0.f, d2 = 0.f;
      if (ed < E) {
        int s = edges[2 * ed];
        g = edges[2 * ed + 1];
        float dx0 = x[3 * s + 0] - x[3 * g + 0];
        float dx1 = x[3 * s + 1] - x[3 * g + 1];
        float dx2 = x[3 * s + 2] - x[3 * g + 2];
        float dd  = sqrtf(dx0 * dx0 + dx1 * dx1 + dx2 * dx2 + 1e-14f);
        float inv = 1.0f / (dd + 1e-5f);
        d0 = dx0 * inv; d1 = dx1 * inv; d2 = dx2 * inv;
      }
      stgt[w][lane] = g;
      sdxn[w][lane * 3 + 0] = d0;
      sdxn[w][lane * 3 + 1] = d1;
      sdxn[w][lane * 3 + 2] = d2;
      svm[w][lane] = 0.0f;
    }
    // comb = (exp(a-max)/(s1+eps)) / (s1/(s1+eps)+eps)
    for (int i = lane; i < 16 * NHD; i += 32) {
      int m = i >> 2, c = i & 3;
      int ed = e0 + m;
      float vv = 0.0f;
      if (ed < E) {
        int g    = edges[2 * ed + 1];
        float a  = attc[(size_t)ed * NHD + c];
        float mx = fdec(amax[g * NHD + c]);
        float s1 = asum[g * NHD + c];
        float ee = __expf(a - mx);
        float a1 = ee / (s1 + 1e-20f);
        float s2 = s1 / (s1 + 1e-20f);
        vv = a1 / (s2 + 1e-20f);
      }
      scomb[w][i] = vv;
    }
    wave_lds_wait();
    // A = h_e_att = outer(he, comb)  (fragment-major), scatter into h_e
    {
      int m = lane & 15;
      int ed = e0 + m;
#pragma unroll
      for (int c = 0; c < 4; ++c) {
        v16h val;
#pragma unroll
        for (int e = 0; e < 16; ++e) {
          int k = a_k_of(lane, c, e);
          float vv = 0.0f;
          if (ed < E) {
            vv = he[(size_t)ed * 32 + (k >> 2)] * scomb[w][m * NHD + (k & 3)];
            atomicAdd(&h_e[(size_t)stgt[w][m] * 128 + k], vv);
          }
          val[e] = (h16)vv;
        }
        frag_st(sA[w], (c << 9) + lane * 16, val);
      }
    }
    wave_lds_wait();
    wave_gemm<128, 128>(sA[w], 0, sWx, lane, [&](int nt, v8f acc) {
      int col = (nt << 4) + nl;
      float wv = svmix[col];
#pragma unroll
      for (int r = 0; r < 8; ++r) {
        int m = mb + r;
        int ed = e0 + m;
        if (ed < E) {
          float cf = tanhf(acc[r]);
          int   g  = stgt[w][m];
          float d0 = sdxn[w][m * 3 + 0];
          float d1 = sdxn[w][m * 3 + 1];
          float d2 = sdxn[w][m * 3 + 2];
          size_t base = ((size_t)g * 128 + col) * 3;
          atomicAdd(&csum[base + 0], cf * d0);
          atomicAdd(&csum[base + 1], cf * d1);
          atomicAdd(&csum[base + 2], cf * d2);
          atomicAdd(&svm[w][m], cf * wv);  // LDS atomic
        }
      }
    });
    wave_lds_wait();
    if (lane < 16) {
      int ed = e0 + lane;
      if (ed < E) {
        float vm = svm[w][lane];
        int   g  = stgt[w][lane];
        atomicAdd(&dvacc[g * 3 + 0], sdxn[w][lane * 3 + 0] * vm);
        atomicAdd(&dvacc[g * 3 + 1], sdxn[w][lane * 3 + 1] * vm);
        atomicAdd(&dvacc[g * 3 + 2], sdxn[w][lane * 3 + 2] * vm);
      }
    }
  }
}

// ---------------------------------------------------------------------------
// Node pass: all node-side MLPs + outputs
__global__ __launch_bounds__(64) void k_node(
    const float* __restrict__ h, const float* __restrict__ x,
    const float* __restrict__ v, const float* W_pn1, const float* b_pn1,
    const float* W_pn2, const float* b_pn2, const float* W_n1,
    const float* b_n1, const float* W_n2, const float* b_n2,
    const float* W_v1, const float* b_v1, const float* w_v2,
    const float* __restrict__ h_e, const float* __restrict__ csum,
    const float* __restrict__ dvacc, const float* __restrict__ cnt,
    float* out_h, float* out_x, float* out_v, int Nn) {
  __shared__ __align__(32) h16 sWpn1[4 * 2 * 512];  // K128 N32
  __shared__ __align__(32) h16 sWpn2[1 * 2 * 512];
  __shared__ __align__(32) h16 sWn1[6 * 2 * 512];   // K192 N32
  __shared__ __align__(32) h16 sWn2[1 * 2 * 512];
  __shared__ __align__(32) h16 sWv1[1 * 2 * 512];
  __shared__ __align__(32) h16 sWv2[1 * 1 * 512];   // col 0 = w_v2
  __shared__ float sb1[32], sb2[32], sb3[32], sb4[32], sb5[32];
  __shared__ __align__(32) h16 sA[2][6 * 512];
  __shared__ __align__(32) h16 sAb[2][512];
  __shared__ __align__(32) h16 sAb2[2][512];

  int tid = threadIdx.x;
  for (int i = tid; i < 128 * 32; i += blockDim.x) {
    int k = i >> 5, n = i & 31;
    sWpn1[b_idx<2>(k, n)] = (h16)W_pn1[i];
  }
  for (int i = tid; i < 32 * 32; i += blockDim.x) {
    int k = i >> 5, n = i & 31;
    sWpn2[b_idx<2>(k, n)] = (h16)W_pn2[i];
    sWn2[b_idx<2>(k, n)]  = (h16)W_n2[i];
    sWv1[b_idx<2>(k, n)]  = (h16)W_v1[i];
  }
  for (int i = tid; i < 192 * 32; i += blockDim.x) {
    int k = i >> 5, n = i & 31;
    sWn1[b_idx<2>(k, n)] = (h16)W_n1[i];
  }
  for (int i = tid; i < 32 * 16; i += blockDim.x) {
    int k = i >> 4, n = i & 15;
    sWv2[b_idx<1>(k, n)] = (h16)((n == 0) ? w_v2[k] : 0.0f);
  }
  for (int i = tid; i < 32; i += blockDim.x) {
    sb1[i] = b_pn1[i]; sb2[i] = b_pn2[i]; sb3[i] = b_n1[i];
    sb4[i] = b_n2[i];  sb5[i] = b_v1[i];
  }
  __syncthreads();

  int w = tid >> 5, lane = tid & 31;
  int nl = lane & 15, mb = (lane & 16) ? 8 : 0;
  int wpb = blockDim.x >> 5;
  int wid = blockIdx.x * wpb + w;
  int wstride = gridDim.x * wpb;
  int tiles = (Nn + 15) >> 4;

  for (int t = wid; t < tiles; t += wstride) {
    int i0 = t << 4;
    // comb_norm -> A chunks 0..3 (fragment-major)
    {
      int m = lane & 15;
      int node = i0 + m;
#pragma unroll
      for (int c = 0; c < 4; ++c) {
        v16h val;
#pragma unroll
        for (int e = 0; e < 16; ++e) {
          int k = a_k_of(lane, c, e);
          float nv = 0.0f;
          if (node < Nn) {
            float cw = cnt[node] + 1e-10f;
            size_t base = ((size_t)node * 128 + k) * 3;
            float a0 = csum[base + 0] / cw;
            float a1 = csum[base + 1] / cw;
            float a2 = csum[base + 2] / cw;
            nv = a0 * a0 + a1 * a1 + a2 * a2;
          }
          val[e] = (h16)nv;
        }
        frag_st(sA[w], (c << 9) + lane * 16, val);
      }
    }
    wave_lds_wait();
    wave_gemm<128, 32>(sA[w], 0, sWpn1, lane, [&](int nt, v8f acc) {
      int n = (nt << 4) + nl;
#pragma unroll
      for (int r = 0; r < 8; ++r)
        sAb[w][a_idx(mb + r, n)] = (h16)siluf(acc[r] + sb1[n]);
    });
    wave_lds_wait();
    wave_gemm<32, 32>(sAb[w], 0, sWpn2, lane, [&](int nt, v8f acc) {
      int n = (nt << 4) + nl;
#pragma unroll
      for (int r = 0; r < 8; ++r)
        sA[w][a_idx(mb + r, 160 + n)] = (h16)siluf(acc[r] + sb2[n]);  // h_comb
    });
    // A chunks 0..4 = [h, h_e]
    {
      int m = lane & 15;
      int node = i0 + m;
#pragma unroll
      for (int c = 0; c < 5; ++c) {
        v16h val;
#pragma unroll
        for (int e = 0; e < 16; ++e) {
          int k = a_k_of(lane, c, e);
          float vv = 0.0f;
          if (node < Nn) {
            vv = (k < 32) ? h[(size_t)node * 32 + k]
                          : h_e[(size_t)node * 128 + (k - 32)];
          }
          val[e] = (h16)vv;
        }
        frag_st(sA[w], (c << 9) + lane * 16, val);
      }
    }
    wave_lds_wait();
    wave_gemm<192, 32>(sA[w], 0, sWn1, lane, [&](int nt, v8f acc) {
      int n = (nt << 4) + nl;
#pragma unroll
      for (int r = 0; r < 8; ++r)
        sAb[w][a_idx(mb + r, n)] = (h16)siluf(acc[r] + sb3[n]);
    });
    wave_lds_wait();
    wave_gemm<32, 32>(sAb[w], 0, sWn2, lane, [&](int nt, v8f acc) {
      int n = (nt << 4) + nl;
#pragma unroll
      for (int r = 0; r < 8; ++r) {
        int m = mb + r;
        int node = i0 + m;
        float hn = 0.0f;
        if (node < Nn) {
          hn = h[(size_t)node * 32 + n] + siluf(acc[r] + sb4[n]);
          out_h[(size_t)node * 32 + n] = hn;
        }
        sAb2[w][a_idx(m, n)] = (h16)hn;
      }
    });
    wave_lds_wait();
    wave_gemm<32, 32>(sAb2[w], 0, sWv1, lane, [&](int nt, v8f acc) {
      int n = (nt << 4) + nl;
#pragma unroll
      for (int r = 0; r < 8; ++r)
        sAb[w][a_idx(mb + r, n)] = (h16)siluf(acc[r] + sb5[n]);
    });
    wave_lds_wait();
    wave_gemm<32, 16>(sAb[w], 0, sWv2, lane, [&](int nt, v8f acc) {
      (void)nt;
      if (nl == 0) {
#pragma unroll
        for (int r = 0; r < 8; ++r) {
          int m = mb + r;
          int node = i0 + m;
          if (node < Nn) {
            float sc = 2.0f / (1.0f + __expf(-acc[r]));
            float cw = cnt[node] + 1e-10f;
#pragma unroll
            for (int kk = 0; kk < 3; ++kk) {
              float dvv = dvacc[node * 3 + kk] / cw;
              float vn  = dvv + sc * v[node * 3 + kk];
              out_v[node * 3 + kk] = vn;
              out_x[node * 3 + kk] = x[node * 3 + kk] + vn;
            }
          }
        }
      }
    });
  }
}

// ---------------------------------------------------------------------------
extern "C" void kernel_launch(void* const* d_in, const int* in_sizes, int n_in,
                              void* d_out, int out_size, void* d_ws,
                              size_t ws_size, hipStream_t stream) {
  (void)n_in; (void)out_size; (void)ws_size;
  const float* h     = (const float*)d_in[0];
  const float* x     = (const float*)d_in[1];
  const float* v     = (const float*)d_in[2];
  const float* W_in  = (const float*)d_in[3];
  const float* b_in  = (const float*)d_in[4];
  const float* W_e1  = (const float*)d_in[5];
  const float* b_e1  = (const float*)d_in[6];
  const float* W_e2  = (const float*)d_in[7];
  const float* b_e2  = (const float*)d_in[8];
  const float* W_att = (const float*)d_in[9];
  const float* b_att = (const float*)d_in[10];
  const float* W_x   = (const float*)d_in[11];
  const float* W_pn1 = (const float*)d_in[12];
  const float* b_pn1 = (const float*)d_in[13];
  const float* W_pn2 = (const float*)d_in[14];
  const float* b_pn2 = (const float*)d_in[15];
  const float* W_n1  = (const float*)d_in[16];
  const float* b_n1  = (const float*)d_in[17];
  const float* W_n2  = (const float*)d_in[18];
  const float* b_n2  = (const float*)d_in[19];
  const float* W_v1  = (const float*)d_in[20];
  const float* b_v1  = (const float*)d_in[21];
  const float* w_v2  = (const float*)d_in[22];
  const float* w_vmix= (const float*)d_in[23];
  const int*   edges = (const int*)d_in[24];

  int E  = in_sizes[24] / 2;
  int Nn = in_sizes[0] / F;

  float* ws = (float*)d_ws;
  size_t off = 0;
  float* he   = ws + off; off += (size_t)E * 32;
  float* attc = ws + off; off += (size_t)E * NHD;
  unsigned* amax = (unsigned*)(ws + off); off += (size_t)Nn * NHD;
  float* zbase = ws + off;
  float* asum  = zbase;
  float* cnt   = asum + (size_t)Nn * NHD;
  float* h_e   = cnt + (size_t)Nn;
  float* csum  = h_e + (size_t)Nn * 128;
  float* dvacc = csum + (size_t)Nn * 384;
  long long zcount = (long long)Nn * (NHD + 1 + 128 + 384 + 3);

  float* out_h = (float*)d_out;
  float* out_x = out_h + (size_t)Nn * 32;
  float* out_v = out_x + (size_t)Nn * 3;

  k_init<<<2048, 256, 0, stream>>>(amax, zbase, zcount, Nn * NHD);
  k_edge1<<<1024, 64, 0, stream>>>(h, x, edges, W_in, b_in, W_e1, b_e1, W_e2,
                                   b_e2, W_att, b_att, he, attc, amax, cnt, E);
  k_edge2<<<(E + 255) / 256, 256, 0, stream>>>(attc, amax, edges, asum, E);
  k_edge3<<<1024, 64, 0, stream>>>(x, edges, he, attc, amax, asum, W_x, w_vmix,
                                   h_e, csum, dvacc, E);
  int tilesN = (Nn + 15) / 16;
  k_node<<<(tilesN + 1) / 2, 64, 0, stream>>>(
      h, x, v, W_pn1, b_pn1, W_pn2, b_pn2, W_n1, b_n1, W_n2, b_n2, W_v1, b_v1,
      w_v2, h_e, csum, dvacc, cnt, out_h, out_x, out_v, Nn);
}
